// CrossAttentionBlock_12317966205103
// MI455X (gfx1250) — matrix operand order repaired and checked by reading, not compiled, not run on hardware
//
#include <hip/hip_runtime.h>
#include <hip/hip_bf16.h>

typedef __bf16 bf16;
typedef __bf16 v16bf __attribute__((ext_vector_type(16)));
typedef __bf16 v8bf  __attribute__((ext_vector_type(8)));
typedef __bf16 v4bf  __attribute__((ext_vector_type(4)));
typedef float  v8f   __attribute__((ext_vector_type(8)));

#define Dm   1024
#define Hh   16
#define HD   64
#define Bb   4
#define Kq   512
#define Nn   4096
#define DFF  4096

static __device__ __forceinline__ v16bf cat8(v8bf lo, v8bf hi) {
  return __builtin_shufflevector(lo, hi, 0,1,2,3,4,5,6,7,8,9,10,11,12,13,14,15);
}

// ---------------------------------------------------------------------------
// Tensor Data Mover: 2D tile (tile_d0 contiguous elems x tile_d1 rows, bf16)
// from global to LDS.  D# packing per CDNA5 ISA.
// This toolchain exposes the 6-arg builtin:
//   (u32x4 g0, i32x8 g1, i32x4 g2, i32x4 g3, i32x8 gx, i32 cpol)
// ---------------------------------------------------------------------------
#if __has_builtin(__builtin_amdgcn_tensor_load_to_lds)
#define HAVE_TDM 1
typedef unsigned int u32x4 __attribute__((ext_vector_type(4)));
typedef int          i32x4 __attribute__((ext_vector_type(4)));
typedef int          i32x8 __attribute__((ext_vector_type(8)));
static __device__ __forceinline__
void tdm_load_2d(unsigned lds_off, const void* gptr,
                 unsigned tensor_d0, unsigned tensor_d1,
                 unsigned tile_d0, unsigned tile_d1, unsigned stride0) {
  unsigned long long ga = (unsigned long long)(uintptr_t)gptr;
  u32x4 g0;
  g0[0] = 1u;                                            // count=1 (user D#)
  g0[1] = lds_off;                                       // lds_addr
  g0[2] = (unsigned)ga;                                  // global_addr[31:0]
  g0[3] = (unsigned)((ga >> 32) & 0x01FFFFFFu) | (2u << 30); // addr[56:32] | type=2
  i32x8 g1;
  g1[0] = 1 << 16;                                       // data_size=1 (2B), mask=0
  g1[1] = (int)((tensor_d0 & 0xFFFFu) << 16);            // tensor_dim0[15:0]
  g1[2] = (int)(((tensor_d0 >> 16) & 0xFFFFu) | ((tensor_d1 & 0xFFFFu) << 16));
  g1[3] = (int)(((tensor_d1 >> 16) & 0xFFFFu) | ((tile_d0 & 0xFFFFu) << 16));
  g1[4] = (int)(tile_d1 & 0xFFFFu);                      // tile_dim1 (tile_dim2=0)
  g1[5] = (int)stride0;                                  // tensor_dim0_stride[31:0]
  g1[6] = 0;
  g1[7] = 0;
  i32x4 z4 = {0, 0, 0, 0};
  i32x8 z8 = {0, 0, 0, 0, 0, 0, 0, 0};
  __builtin_amdgcn_tensor_load_to_lds(g0, g1, z4, z4, z8, 0);
}
#endif

// ---------------------------------------------------------------------------
// Weight transpose + fp32 -> bf16 cast:  Wt[n*Kd + k] = (bf16) W[k*Nc + n]
// ---------------------------------------------------------------------------
__global__ void transpose_cvt_kernel(const float* __restrict__ W, bf16* __restrict__ Wt,
                                     int Kd, int Nc) {
  size_t o = (size_t)blockIdx.x * blockDim.x + threadIdx.x;
  if (o >= (size_t)Kd * Nc) return;
  int n = (int)(o / Kd);
  int k = (int)(o % Kd);
  Wt[o] = (bf16)W[(size_t)k * Nc + n];
}

// ---------------------------------------------------------------------------
// LayerNorm (wave per row of 1024) -> bf16
// ---------------------------------------------------------------------------
__global__ __launch_bounds__(256)
void layernorm_kernel(const float* __restrict__ x, const float* __restrict__ g,
                      const float* __restrict__ b, bf16* __restrict__ out, int rows) {
  int wave = (int)(((size_t)blockIdx.x * blockDim.x + threadIdx.x) >> 5);
  int lane = threadIdx.x & 31;
  if (wave >= rows) return;
  const float4* xr = (const float4*)(x + (size_t)wave * Dm);
  float4 v[8];
  float s = 0.f;
  #pragma unroll
  for (int i = 0; i < 8; ++i) {
    v[i] = xr[lane + i * 32];
    s += v[i].x + v[i].y + v[i].z + v[i].w;
  }
  #pragma unroll
  for (int d = 16; d >= 1; d >>= 1) s += __shfl_xor(s, d, 32);
  float mu = s * (1.f / Dm);
  float var = 0.f;
  #pragma unroll
  for (int i = 0; i < 8; ++i) {
    float dx = v[i].x - mu, dy = v[i].y - mu, dz = v[i].z - mu, dw = v[i].w - mu;
    var += dx * dx + dy * dy + dz * dz + dw * dw;
  }
  #pragma unroll
  for (int d = 16; d >= 1; d >>= 1) var += __shfl_xor(var, d, 32);
  float inv = rsqrtf(var * (1.f / Dm) + 1e-5f);
  v4bf* orow = (v4bf*)(out + (size_t)wave * Dm);
  #pragma unroll
  for (int i = 0; i < 8; ++i) {
    int c = (lane + i * 32) * 4;
    v4bf o;
    o[0] = (bf16)((v[i].x - mu) * inv * g[c + 0] + b[c + 0]);
    o[1] = (bf16)((v[i].y - mu) * inv * g[c + 1] + b[c + 1]);
    o[2] = (bf16)((v[i].z - mu) * inv * g[c + 2] + b[c + 2]);
    o[3] = (bf16)((v[i].w - mu) * inv * g[c + 3] + b[c + 3]);
    orow[lane + i * 32] = o;
  }
}

// ---------------------------------------------------------------------------
// RMSNorm (wave per row of 1024), fp32 in -> bf16 out
// ---------------------------------------------------------------------------
__global__ __launch_bounds__(256)
void rmsnorm_kernel(const float* __restrict__ x, const float* __restrict__ w,
                    bf16* __restrict__ out, int rows) {
  int wave = (int)(((size_t)blockIdx.x * blockDim.x + threadIdx.x) >> 5);
  int lane = threadIdx.x & 31;
  if (wave >= rows) return;
  const float4* xr = (const float4*)(x + (size_t)wave * Dm);
  float4 v[8];
  float ms = 0.f;
  #pragma unroll
  for (int i = 0; i < 8; ++i) {
    v[i] = xr[lane + i * 32];
    ms += v[i].x * v[i].x + v[i].y * v[i].y + v[i].z * v[i].z + v[i].w * v[i].w;
  }
  #pragma unroll
  for (int d = 16; d >= 1; d >>= 1) ms += __shfl_xor(ms, d, 32);
  float inv = rsqrtf(ms * (1.f / Dm) + 1e-6f);
  v4bf* orow = (v4bf*)(out + (size_t)wave * Dm);
  #pragma unroll
  for (int i = 0; i < 8; ++i) {
    int c = (lane + i * 32) * 4;
    v4bf o;
    o[0] = (bf16)(v[i].x * inv * w[c + 0]);
    o[1] = (bf16)(v[i].y * inv * w[c + 1]);
    o[2] = (bf16)(v[i].z * inv * w[c + 2]);
    o[3] = (bf16)(v[i].w * inv * w[c + 3]);
    orow[lane + i * 32] = o;
  }
}

// ---------------------------------------------------------------------------
// Pack V:  Vt[b,h,hd,n] (bf16) = v_lin[b,n,h*64+hd] (f32)
// ---------------------------------------------------------------------------
__global__ void pack_vt_kernel(const float* __restrict__ v, bf16* __restrict__ vt) {
  size_t i = (size_t)blockIdx.x * blockDim.x + threadIdx.x;
  if (i >= (size_t)Bb * Nn * Dm) return;
  int d = (int)(i & (Dm - 1));
  size_t bn = i >> 10;
  int n = (int)(bn & (Nn - 1));
  int b = (int)(bn >> 12);
  int h = d >> 6, hd = d & 63;
  vt[(((size_t)((b * Hh + h) * HD + hd)) << 12) + n] = (bf16)v[i];
}

// ---------------------------------------------------------------------------
// Block-cooperative WMMA GEMM: 128x128 block tile, double-buffered LDS
// staged by the Tensor Data Mover (DMA of stage k+1 overlaps WMMA of stage k).
// 8 waves; each wave computes 32x64 (8x v_wmma_f32_16x16x32_bf16 per K-step).
// ---------------------------------------------------------------------------
#define EPI_RAW   0   // outF = acc + bias
#define EPI_RESID 1   // outF = acc + bias + resid
#define EPI_GELU  2   // outB = bf16(gelu(acc + bias))   (exact erf gelu)
#define EPI_ADD   3   // outF += acc + bias

template <int EPI>
__global__ __launch_bounds__(256)
void gemm_wmma_kernel(const bf16* __restrict__ A, const bf16* __restrict__ Wt,
                      const float* __restrict__ bias, const float* __restrict__ resid,
                      float* __restrict__ outF, bf16* __restrict__ outB,
                      int M, int Kd, int Nc) {
  __shared__ __align__(128) bf16 sA[2][128 * 32];   // 128 M-rows x 32 K, x2 stages
  __shared__ __align__(128) bf16 sB[2][128 * 32];   // 128 N-rows x 32 K, x2 stages
  int lane = threadIdx.x & 31;
  int wib  = threadIdx.x >> 5;       // wave in block, 0..7
  int wm   = wib & 3;                // 4 waves along M  (32 rows each)
  int wn   = wib >> 2;               // 2 waves along N  (64 cols each)
  int nBlk = Nc >> 7;
  int tm = blockIdx.x / nBlk, tn = blockIdx.x % nBlk;
  int ma = lane & 15, kg = lane >> 4, nb = lane & 15;

  const bf16* Abase = A + (size_t)tm * 128 * Kd;
  const bf16* Bbase = Wt + (size_t)tn * 128 * Kd;
  int steps = Kd >> 5;

#if HAVE_TDM
  if (wib == 0) {   // prologue: DMA stage 0
    tdm_load_2d((unsigned)(uintptr_t)&sA[0][0], Abase, Kd, M,  32, 128, Kd);
    tdm_load_2d((unsigned)(uintptr_t)&sB[0][0], Bbase, Kd, Nc, 32, 128, Kd);
  }
#endif

  v8f acc[2][4] = {};
  for (int ks = 0; ks < steps; ++ks) {
    int cur = ks & 1;
    __syncthreads();                       // all waves done reading stage ks-1
#if HAVE_TDM
    if (wib == 0) {
      if (ks + 1 < steps) {
        int nxt = (ks + 1) & 1;
        int kk = (ks + 1) << 5;
        tdm_load_2d((unsigned)(uintptr_t)&sA[nxt][0], Abase + kk, Kd, M,  32, 128, Kd);
        tdm_load_2d((unsigned)(uintptr_t)&sB[nxt][0], Bbase + kk, Kd, Nc, 32, 128, Kd);
        // TDM ops complete in order: <=2 outstanding => stage ks has landed,
        // only stage ks+1's two loads remain in flight.
        __builtin_amdgcn_s_wait_tensorcnt(2);
      } else {
        __builtin_amdgcn_s_wait_tensorcnt(0);
      }
    }
#else
    {
      int kk = ks << 5;
      for (int c = threadIdx.x; c < 512; c += 256) {
        int row = c >> 2, xo = (c & 3) * 8;
        *(v8bf*)&sA[cur][row * 32 + xo] = *(const v8bf*)(Abase + kk + (size_t)row * Kd + xo);
        *(v8bf*)&sB[cur][row * 32 + xo] = *(const v8bf*)(Bbase + kk + (size_t)row * Kd + xo);
      }
    }
#endif
    __syncthreads();                       // stage ks visible to all waves
    #pragma unroll
    for (int half = 0; half < 2; ++half) {
      const bf16* ar = &sA[cur][(wm * 32 + half * 16 + ma) * 32 + kg * 8];
      v16bf a = cat8(*(const v8bf*)(ar), *(const v8bf*)(ar + 16));
      #pragma unroll
      for (int s = 0; s < 4; ++s) {
        const bf16* br = &sB[cur][(wn * 64 + s * 16 + nb) * 32 + kg * 8];
        v16bf b = cat8(*(const v8bf*)(br), *(const v8bf*)(br + 16));
        acc[half][s] = __builtin_amdgcn_wmma_f32_16x16x32_bf16(
            false, a, false, b, (short)0, acc[half][s], false, false);
      }
    }
  }

  #pragma unroll
  for (int half = 0; half < 2; ++half) {
    #pragma unroll
    for (int s = 0; s < 4; ++s) {
      int col = tn * 128 + wn * 64 + s * 16 + nb;
      float bs = bias[col];
      #pragma unroll
      for (int r = 0; r < 8; ++r) {
        int row = tm * 128 + wm * 32 + half * 16 + r + kg * 8;
        size_t idx = (size_t)row * Nc + col;
        float v = acc[half][s][r] + bs;
        if (EPI == EPI_RAW)        outF[idx] = v;
        else if (EPI == EPI_RESID) outF[idx] = v + resid[idx];
        else if (EPI == EPI_ADD)   outF[idx] += v;
        else                       outB[idx] = (bf16)(0.5f * v * (1.f + erff(v * 0.70710678118654752f)));
      }
    }
  }
}

// ---------------------------------------------------------------------------
// Flash-style attention: wave owns (b, h, 16 q-rows); online softmax,
// streaming 64 key columns per step (16 WMMAs / chunk).
// ---------------------------------------------------------------------------
__global__ __launch_bounds__(256)
void attn_kernel(const bf16* __restrict__ q, const bf16* __restrict__ k,
                 const bf16* __restrict__ vt, bf16* __restrict__ ctx) {
  __shared__ __align__(32) bf16 pbuf[8][16 * 64];
  int wv = (int)(((size_t)blockIdx.x * blockDim.x + threadIdx.x) >> 5);
  if (wv >= Bb * Hh * (Kq / 16)) return;
  int winb = threadIdx.x >> 5;
  int lane = threadIdx.x & 31;
  int qt = wv & 31;
  int h  = (wv >> 5) & 15;
  int b  = wv >> 9;
  int ma = lane & 15, kg = lane >> 4, nb = lane & 15;

  const bf16* qrow = q + (size_t)(b * Kq + qt * 16 + ma) * Dm + h * HD + kg * 8;
  v16bf aq0 = cat8(*(const v8bf*)(qrow),      *(const v8bf*)(qrow + 16));
  v16bf aq1 = cat8(*(const v8bf*)(qrow + 32), *(const v8bf*)(qrow + 48));

  float mstate[8], lstate[8];
  #pragma unroll
  for (int r = 0; r < 8; ++r) { mstate[r] = -3.0e38f; lstate[r] = 0.f; }
  v8f o[4] = {};

  const bf16* kbase = k + ((size_t)b * Nn) * Dm + h * HD + kg * 8;
  const bf16* vbase = vt + (size_t)((b * Hh + h) * HD + nb) * Nn + kg * 8;
  bf16* pb = pbuf[winb];

  for (int nc = 0; nc < Nn; nc += 64) {
    // S = Q K^T : four 16x16 column tiles, HD split into 2 WMMA k-steps
    v8f sc[4];
    #pragma unroll
    for (int j = 0; j < 4; ++j) {
      const bf16* kr = kbase + (size_t)(nc + j * 16 + nb) * Dm;
      v8f z = {};
      sc[j] = __builtin_amdgcn_wmma_f32_16x16x32_bf16(false, aq0, false, cat8(*(const v8bf*)(kr), *(const v8bf*)(kr + 16)),      (short)0, z,     false, false);
      sc[j] = __builtin_amdgcn_wmma_f32_16x16x32_bf16(false, aq1, false, cat8(*(const v8bf*)(kr + 32), *(const v8bf*)(kr + 48)), (short)0, sc[j], false, false);
    }
    if (nc + 64 < Nn) {
      __builtin_prefetch(kbase + (size_t)(nc + 64 + nb) * Dm, 0, 1);
      __builtin_prefetch(vbase + nc + 64, 0, 1);
    }

    // online softmax across the 64-column chunk
    float p[4][8];
    #pragma unroll
    for (int r = 0; r < 8; ++r) {
      float x0 = sc[0][r] * 0.125f, x1 = sc[1][r] * 0.125f;
      float x2 = sc[2][r] * 0.125f, x3 = sc[3][r] * 0.125f;
      float mx = fmaxf(fmaxf(x0, x1), fmaxf(x2, x3));
      #pragma unroll
      for (int d = 8; d >= 1; d >>= 1) mx = fmaxf(mx, __shfl_xor(mx, d, 32));
      float nm = fmaxf(mstate[r], mx);
      float e0 = __expf(x0 - nm), e1 = __expf(x1 - nm);
      float e2 = __expf(x2 - nm), e3 = __expf(x3 - nm);
      float sum = (e0 + e1) + (e2 + e3);
      #pragma unroll
      for (int d = 8; d >= 1; d >>= 1) sum += __shfl_xor(sum, d, 32);
      float corr = __expf(mstate[r] - nm);
      lstate[r] = lstate[r] * corr + sum;
      mstate[r] = nm;
      #pragma unroll
      for (int t = 0; t < 4; ++t) o[t][r] *= corr;
      p[0][r] = e0; p[1][r] = e1; p[2][r] = e2; p[3][r] = e3;
    }

    // C-layout -> A-layout transpose of P through LDS (same-wave DS in-order)
    #pragma unroll
    for (int r = 0; r < 8; ++r) {
      int m = r + kg * 8;
      #pragma unroll
      for (int j = 0; j < 4; ++j) pb[m * 64 + j * 16 + nb] = (bf16)p[j][r];
    }
    asm volatile("s_wait_dscnt 0" ::: "memory");
    v16bf pa0 = cat8(*(const v8bf*)(pb + ma * 64 + kg * 8),
                     *(const v8bf*)(pb + ma * 64 + 16 + kg * 8));
    v16bf pa1 = cat8(*(const v8bf*)(pb + ma * 64 + 32 + kg * 8),
                     *(const v8bf*)(pb + ma * 64 + 48 + kg * 8));

    // O += P V : 4 hd-column tiles, 64 tokens = 2 chained WMMAs each
    #pragma unroll
    for (int t = 0; t < 4; ++t) {
      const bf16* vr = vbase + (size_t)t * 16 * Nn + nc;
      o[t] = __builtin_amdgcn_wmma_f32_16x16x32_bf16(false, pa0, false, cat8(*(const v8bf*)(vr), *(const v8bf*)(vr + 16)),      (short)0, o[t], false, false);
      o[t] = __builtin_amdgcn_wmma_f32_16x16x32_bf16(false, pa1, false, cat8(*(const v8bf*)(vr + 32), *(const v8bf*)(vr + 48)), (short)0, o[t], false, false);
    }
  }

  bf16* crow = ctx + (size_t)(b * Kq + qt * 16) * Dm + h * HD;
  #pragma unroll
  for (int t = 0; t < 4; ++t) {
    int col = t * 16 + nb;
    #pragma unroll
    for (int r = 0; r < 8; ++r) {
      int m = r + kg * 8;
      crow[(size_t)m * Dm + col] = (bf16)(o[t][r] / lstate[r]);
    }
  }
}

// ---------------------------------------------------------------------------
// Host-side orchestration
// ---------------------------------------------------------------------------
#define MiB (1024ull * 1024ull)

extern "C" void kernel_launch(void* const* d_in, const int* in_sizes, int n_in,
                              void* d_out, int out_size, void* d_ws, size_t ws_size,
                              hipStream_t stream) {
  const float* query  = (const float*)d_in[0];
  const float* points = (const float*)d_in[1];
  const float* ln_q_g = (const float*)d_in[2];
  const float* ln_q_b = (const float*)d_in[3];
  const float* ln_kv_g= (const float*)d_in[4];
  const float* ln_kv_b= (const float*)d_in[5];
  const float* Wq = (const float*)d_in[6];   const float* bq = (const float*)d_in[7];
  const float* Wk = (const float*)d_in[8];   const float* bk = (const float*)d_in[9];
  const float* Wv = (const float*)d_in[10];  const float* bv = (const float*)d_in[11];
  const float* rms_q_w = (const float*)d_in[12];
  const float* rms_k_w = (const float*)d_in[13];
  const float* Wo = (const float*)d_in[14];  const float* bo = (const float*)d_in[15];
  const float* ln_mlp_g = (const float*)d_in[16];
  const float* ln_mlp_b = (const float*)d_in[17];
  const float* W1 = (const float*)d_in[18];  const float* b1 = (const float*)d_in[19];
  const float* W2 = (const float*)d_in[20];  const float* b2 = (const float*)d_in[21];
  float* out = (float*)d_out;

  char* ws = (char*)d_ws;
  bf16*  q_ln  = (bf16*) (ws + 0);            //  4 MiB  [2048,1024]
  bf16*  kv_ln = (bf16*) (ws + 4   * MiB);    // 32 MiB  [16384,1024]
  bf16*  WqT   = (bf16*) (ws + 36  * MiB);    //  2 MiB  [1024,1024]
  bf16*  WkT   = (bf16*) (ws + 38  * MiB);
  bf16*  WvT   = (bf16*) (ws + 40  * MiB);
  bf16*  WoT   = (bf16*) (ws + 42  * MiB);
  bf16*  W1T   = (bf16*) (ws + 44  * MiB);    //  8 MiB  [4096,1024]
  bf16*  W2T   = (bf16*) (ws + 52  * MiB);    //  8 MiB  [1024,4096]
  float* lin   = (float*)(ws + 60  * MiB);    // 64 MiB  shared f32 GEMM out
  bf16*  qb    = (bf16*) (ws + 124 * MiB);    //  4 MiB
  bf16*  kb    = (bf16*) (ws + 128 * MiB);    // 32 MiB
  bf16*  vtb   = (bf16*) (ws + 160 * MiB);    // 32 MiB  [B,H,HD,N]
  bf16*  ctx   = (bf16*) (ws + 192 * MiB);    //  4 MiB
  bf16*  hmlp  = (bf16*) (ws + 196 * MiB);    //  4 MiB
  bf16*  mlp1  = (bf16*) (ws + 200 * MiB);    // 16 MiB  [2048,4096]

  const int BK = Bb * Kq;   // 2048
  const int BN = Bb * Nn;   // 16384
  dim3 blk(256);

  // 1) pack weights -> transposed bf16
  transpose_cvt_kernel<<<(Dm * Dm) / 256, blk, 0, stream>>>(Wq, WqT, Dm, Dm);
  transpose_cvt_kernel<<<(Dm * Dm) / 256, blk, 0, stream>>>(Wk, WkT, Dm, Dm);
  transpose_cvt_kernel<<<(Dm * Dm) / 256, blk, 0, stream>>>(Wv, WvT, Dm, Dm);
  transpose_cvt_kernel<<<(Dm * Dm) / 256, blk, 0, stream>>>(Wo, WoT, Dm, Dm);
  transpose_cvt_kernel<<<(Dm * DFF) / 256, blk, 0, stream>>>(W1, W1T, Dm, DFF);
  transpose_cvt_kernel<<<(DFF * Dm) / 256, blk, 0, stream>>>(W2, W2T, DFF, Dm);

  // 2) LayerNorms -> bf16
  layernorm_kernel<<<BK / 8, blk, 0, stream>>>(query, ln_q_g, ln_q_b, q_ln, BK);
  layernorm_kernel<<<BN / 8, blk, 0, stream>>>(points, ln_kv_g, ln_kv_b, kv_ln, BN);

  // 3) Q projection + RMSNorm
  gemm_wmma_kernel<EPI_RAW><<<(BK / 128) * (Dm / 128), blk, 0, stream>>>(
      q_ln, WqT, bq, nullptr, lin, nullptr, BK, Dm, Dm);
  rmsnorm_kernel<<<BK / 8, blk, 0, stream>>>(lin, rms_q_w, qb, BK);

  // 4) K projection + RMSNorm
  gemm_wmma_kernel<EPI_RAW><<<(BN / 128) * (Dm / 128), blk, 0, stream>>>(
      kv_ln, WkT, bk, nullptr, lin, nullptr, BN, Dm, Dm);
  rmsnorm_kernel<<<BN / 8, blk, 0, stream>>>(lin, rms_k_w, kb, BN);

  // 5) V projection + transpose pack
  gemm_wmma_kernel<EPI_RAW><<<(BN / 128) * (Dm / 128), blk, 0, stream>>>(
      kv_ln, WvT, bv, nullptr, lin, nullptr, BN, Dm, Dm);
  pack_vt_kernel<<<(int)(((size_t)Bb * Nn * Dm) / 256), blk, 0, stream>>>(lin, vtb);

  // 6) attention -> ctx (bf16)
  attn_kernel<<<(Bb * Hh * (Kq / 16)) / 8, blk, 0, stream>>>(qb, kb, vtb, ctx);

  // 7) out = ctx @ Wo + bo + query   (into d_out; doubles as residual)
  gemm_wmma_kernel<EPI_RESID><<<(BK / 128) * (Dm / 128), blk, 0, stream>>>(
      ctx, WoT, bo, query, out, nullptr, BK, Dm, Dm);

  // 8) MLP
  layernorm_kernel<<<BK / 8, blk, 0, stream>>>(out, ln_mlp_g, ln_mlp_b, hmlp, BK);
  gemm_wmma_kernel<EPI_GELU><<<(BK / 128) * (DFF / 128), blk, 0, stream>>>(
      hmlp, W1T, b1, nullptr, nullptr, mlp1, BK, Dm, DFF);
  gemm_wmma_kernel<EPI_ADD><<<(BK / 128) * (Dm / 128), blk, 0, stream>>>(
      mlp1, W2T, b2, nullptr, out, nullptr, BK, DFF, Dm);

  (void)in_sizes; (void)n_in; (void)out_size; (void)ws_size;
}
